// Model_24300924961064
// MI455X (gfx1250) — matrix operand order repaired and checked by reading, not compiled, not run on hardware
//
#include <hip/hip_runtime.h>
#include <hip/hip_bf16.h>

// ---------------------------------------------------------------------------
// Problem constants (from the reference)
// ---------------------------------------------------------------------------
#define PDIM   168
#define HIDC   32
#define CK     6
#define HIDR   100
#define HIDS   5
#define SKIP   24
#define HW     24
#define BATCH  128
#define LLEN   163            // P - CK + 1
#define PT     6              // (P - CK) / SKIP
#define MOUT   24             // M1*M2*M3
#define KDIM   (PDIM * MOUT)  // 4032
#define ODIM   (PDIM * HIDC * CK) // 32256

typedef __attribute__((ext_vector_type(16))) _Float16 v16h;
typedef __attribute__((ext_vector_type(8)))  _Float16 v8h;
typedef __attribute__((ext_vector_type(8)))  float    v8f;

__device__ __forceinline__ float sigf(float x) {
    return 1.0f / (1.0f + __expf(-x));
}

// ---------------------------------------------------------------------------
// Kernel 1: conv einsum as GEMM  C1[128][32256] = relu(X[128][4032] * W^T + b)
// f32 -> f16 conversion at stage time; f32 accumulation in WMMA.
// Block = 128x128 output tile, 8 waves; wave w owns M-tile w (8 N-tiles).
// W is staged 128 cols x 64 k per iteration into double-buffered LDS (f16),
// one barrier per iteration; next chunk's global loads overlap the WMMAs.
// B-fragment LDS loads are software-pipelined one tile ahead of each WMMA.
// ---------------------------------------------------------------------------
__global__ __launch_bounds__(256) void conv_gemm_kernel(
        const float* __restrict__ x,      // [128][4032]
        const float* __restrict__ w,      // [32256][4032]
        const float* __restrict__ bias,   // [32256]
        float* __restrict__ C1)           // [128][32256]
{
    __shared__ __align__(32) _Float16 Bbuf[2][128 * 64]; // [buf][col][k] f16, 2x16KB

    const int tid  = threadIdx.x;
    const int wv   = tid >> 5;
    const int lane = tid & 31;
    const int lm   = lane & 15;
    const bool hiL = lane >= 16;
    const int nbase = blockIdx.x * 128;

    v8f acc[8];
    {
        v8f z = {};
        #pragma unroll
        for (int i = 0; i < 8; ++i) acc[i] = z;
    }

    const int m = wv * 16 + lm;                       // A row (0..127)
    const float* xrow = x + (size_t)m * KDIM;
    const int khA = hiL ? 16 : 0;                     // A-operand K half
    const int khB = hiL ? 16 : 0;                     // B-operand K half

    // staging geometry: thread -> (col = tid>>1, 16-wide half = (tid&1)*16)
    const int scol  = tid >> 1;
    const int shalf = (tid & 1) * 16;
    const float* sbase = w + (size_t)(nbase + scol) * KDIM + shalf;
    _Float16* dbase = &Bbuf[0][0] + scol * 64 + shalf;

    auto stage = [&](int buf, int kbn) {
        const float* src0 = sbase + kbn;          // K seg [shalf, shalf+16)
        const float* src1 = sbase + kbn + 32;     // K seg [32+shalf, 32+shalf+16)
        float4 f0 = ((const float4*)src0)[0];
        float4 f1 = ((const float4*)src0)[1];
        float4 f2 = ((const float4*)src0)[2];
        float4 f3 = ((const float4*)src0)[3];
        float4 g0 = ((const float4*)src1)[0];
        float4 g1 = ((const float4*)src1)[1];
        float4 g2 = ((const float4*)src1)[2];
        float4 g3 = ((const float4*)src1)[3];
        _Float16* d0 = dbase + buf * (128 * 64);
        d0[0]  = (_Float16)f0.x; d0[1]  = (_Float16)f0.y;
        d0[2]  = (_Float16)f0.z; d0[3]  = (_Float16)f0.w;
        d0[4]  = (_Float16)f1.x; d0[5]  = (_Float16)f1.y;
        d0[6]  = (_Float16)f1.z; d0[7]  = (_Float16)f1.w;
        d0[8]  = (_Float16)f2.x; d0[9]  = (_Float16)f2.y;
        d0[10] = (_Float16)f2.z; d0[11] = (_Float16)f2.w;
        d0[12] = (_Float16)f3.x; d0[13] = (_Float16)f3.y;
        d0[14] = (_Float16)f3.z; d0[15] = (_Float16)f3.w;
        _Float16* d1 = d0 + 32;
        d1[0]  = (_Float16)g0.x; d1[1]  = (_Float16)g0.y;
        d1[2]  = (_Float16)g0.z; d1[3]  = (_Float16)g0.w;
        d1[4]  = (_Float16)g1.x; d1[5]  = (_Float16)g1.y;
        d1[6]  = (_Float16)g1.z; d1[7]  = (_Float16)g1.w;
        d1[8]  = (_Float16)g2.x; d1[9]  = (_Float16)g2.y;
        d1[10] = (_Float16)g2.z; d1[11] = (_Float16)g2.w;
        d1[12] = (_Float16)g3.x; d1[13] = (_Float16)g3.y;
        d1[14] = (_Float16)g3.z; d1[15] = (_Float16)g3.w;
        if (kbn + 64 < KDIM) __builtin_prefetch(src0 + 64, 0, 3);
    };

    stage(0, 0);
    __syncthreads();

    int p = 0;
    for (int kb = 0; kb < KDIM; kb += 64) {
        // ---- issue next chunk's staging first (overlaps the WMMAs) -------
        if (kb + 64 < KDIM) stage(p ^ 1, kb + 64);

        // ---- two K=32 sub-chunks against the ready buffer ----------------
        #pragma unroll
        for (int kc = 0; kc < 2; ++kc) {
            // A fragment (per-wave, from global; x is tiny & L2-hot)
            v16h a;
            {
                const float* ap = xrow + kb + kc * 32 + khA;
                float4 a0 = ((const float4*)ap)[0];
                float4 a1 = ((const float4*)ap)[1];
                float4 a2 = ((const float4*)(ap + 8))[0];
                float4 a3 = ((const float4*)(ap + 8))[1];
                a[0]  = (_Float16)a0.x; a[1]  = (_Float16)a0.y;
                a[2]  = (_Float16)a0.z; a[3]  = (_Float16)a0.w;
                a[4]  = (_Float16)a1.x; a[5]  = (_Float16)a1.y;
                a[6]  = (_Float16)a1.z; a[7]  = (_Float16)a1.w;
                a[8]  = (_Float16)a2.x; a[9]  = (_Float16)a2.y;
                a[10] = (_Float16)a2.z; a[11] = (_Float16)a2.w;
                a[12] = (_Float16)a3.x; a[13] = (_Float16)a3.y;
                a[14] = (_Float16)a3.z; a[15] = (_Float16)a3.w;
            }
            // software-pipeline B-fragment loads one tile ahead of each WMMA
            const _Float16* bb = &Bbuf[p][0] + kc * 32 + khB;
            v16h bcur = *(const v16h*)&bb[(0 * 16 + lm) * 64];
            #pragma unroll
            for (int nt = 0; nt < 8; ++nt) {
                v16h bnxt = bcur;
                if (nt < 7) bnxt = *(const v16h*)&bb[((nt + 1) * 16 + lm) * 64];
                acc[nt] = __builtin_amdgcn_wmma_f32_16x16x32_f16(
                    false, a, false, bcur, (short)0, acc[nt], false, false);
                bcur = bnxt;
            }
        }
        __syncthreads();
        p ^= 1;
    }

    // ---- epilogue: bias + relu + store -----------------------------------
    #pragma unroll
    for (int nt = 0; nt < 8; ++nt) {
        const int n = nbase + nt * 16 + lm;
        const float bv = bias[n];
        #pragma unroll
        for (int r = 0; r < 8; ++r) {
            const int row = wv * 16 + (hiL ? 8 : 0) + r;
            float v = acc[nt][r] + bv;
            C1[(size_t)row * ODIM + n] = v > 0.0f ? v : 0.0f;
        }
    }
}

// ---------------------------------------------------------------------------
// Kernel 2: sliding diagonal window sum
// c2[b][hc][l] = sum_ck C1[b][(hc*6+ck)*168 + ck + l]
// ---------------------------------------------------------------------------
__global__ __launch_bounds__(256) void window_sum_kernel(
        const float* __restrict__ C1, float* __restrict__ c2)
{
    const int idx = blockIdx.x * 256 + threadIdx.x;
    if (idx >= BATCH * HIDC * LLEN) return;
    const int l   = idx % LLEN;
    const int hc  = (idx / LLEN) % HIDC;
    const int b   = idx / (LLEN * HIDC);
    const float* base = C1 + (size_t)b * ODIM;
    float s = 0.0f;
    #pragma unroll
    for (int ck = 0; ck < CK; ++ck)
        s += base[(hc * CK + ck) * PDIM + ck + l];
    c2[idx] = s;
}

// ---------------------------------------------------------------------------
// Kernel 3: precompute GRU1 input gates  gi1[t][b][g] = c2seq . Wih^T + bih
// ---------------------------------------------------------------------------
__global__ __launch_bounds__(256) void gi1_kernel(
        const float* __restrict__ c2,    // [128][32][163]
        const float* __restrict__ Wih,   // [300][32]
        const float* __restrict__ bih,   // [300]
        float* __restrict__ gi1)         // [163][128][300]
{
    const int idx = blockIdx.x * 256 + threadIdx.x;
    if (idx >= LLEN * BATCH * 300) return;
    const int g = idx % 300;
    const int b = (idx / 300) % BATCH;
    const int t = idx / (300 * BATCH);
    float s = bih[g];
    const float* wr = Wih + g * HIDC;
    #pragma unroll 8
    for (int i = 0; i < HIDC; ++i)
        s += c2[((size_t)b * HIDC + i) * LLEN + t] * wr[i];
    gi1[idx] = s;
}

// ---------------------------------------------------------------------------
// Kernel 4: precompute GRU-skip input gates  gis[pt][row][g]
// ---------------------------------------------------------------------------
__global__ __launch_bounds__(256) void gis_kernel(
        const float* __restrict__ c2,     // [128][32][163]
        const float* __restrict__ Wih,    // [15][32]
        const float* __restrict__ bih,    // [15]
        float* __restrict__ gis)          // [6][3072][15]
{
    const int idx = blockIdx.x * 256 + threadIdx.x;
    if (idx >= PT * BATCH * SKIP * 15) return;
    const int g   = idx % 15;
    const int row = (idx / 15) % (BATCH * SKIP);
    const int pt  = idx / (15 * BATCH * SKIP);
    const int b   = row / SKIP;
    const int sk  = row % SKIP;
    const int l   = (LLEN - PT * SKIP) + pt * SKIP + sk; // 19 + ...
    float s = bih[g];
    const float* wr = Wih + g * HIDC;
    #pragma unroll 8
    for (int i = 0; i < HIDC; ++i)
        s += c2[((size_t)b * HIDC + i) * LLEN + l] * wr[i];
    gis[idx] = s;
}

// ---------------------------------------------------------------------------
// Kernel 5: GRU1 recurrence with WMMA.
// 8 blocks x 16 batch rows.  Whh lives in B-operand registers for all 163
// steps; per step only h (16x100) is rebuilt as f16 A fragments from LDS.
// ---------------------------------------------------------------------------
__global__ __launch_bounds__(256) void gru1_kernel(
        const float* __restrict__ gi1,   // [163][128][300] (includes bih)
        const float* __restrict__ Whh,   // [300][100]
        const float* __restrict__ bhh,   // [300]
        float* __restrict__ h_out)       // [128][100]
{
    __shared__ float hbuf[16 * HIDR];                  // h (f32)
    __shared__ float ghbuf[16 * 300];                  // h @ Whh^T
    __shared__ __align__(32) _Float16 hf16[16 * 128];  // h (f16, K padded to 128)

    const int tid  = threadIdx.x;
    const int wv   = tid >> 5;
    const int lane = tid & 31;
    const int lm   = lane & 15;
    const bool hiL = lane >= 16;
    const int bglob0 = blockIdx.x * 16;

    const int nt0 = wv, nt1 = wv + 8, nt2 = wv + 16;
    const bool has2 = (nt2 < 19);                      // wave-uniform

    // ---- load Whh into resident B fragments (zero-padded to K=128) -------
    v16h B0[4], B1[4], B2[4];
    {
        const int khB = hiL ? 16 : 0;
        #pragma unroll
        for (int kc = 0; kc < 4; ++kc) {
            v16h b0, b1, b2;
            #pragma unroll
            for (int e = 0; e < 16; ++e) {
                const int K = kc * 32 + khB + e;
                const int j0 = nt0 * 16 + lm;
                const int j1 = nt1 * 16 + lm;
                const int j2 = nt2 * 16 + lm;
                b0[e] = (K < HIDR)              ? (_Float16)Whh[j0 * HIDR + K] : (_Float16)0.0f;
                b1[e] = (K < HIDR)              ? (_Float16)Whh[j1 * HIDR + K] : (_Float16)0.0f;
                b2[e] = (K < HIDR && j2 < 300)  ? (_Float16)Whh[j2 * HIDR + K] : (_Float16)0.0f;
            }
            B0[kc] = b0; B1[kc] = b1; B2[kc] = b2;
        }
    }

    // ---- init h = 0 -------------------------------------------------------
    for (int i = tid; i < 16 * HIDR; i += 256) hbuf[i] = 0.0f;
    for (int i = tid; i < 16 * 128;  i += 256) hf16[i] = (_Float16)0.0f;
    __syncthreads();

    const int khA = hiL ? 16 : 0;

    for (int t = 0; t < LLEN; ++t) {
        // ---- build A fragments from hf16 in LDS --------------------------
        v16h A[4];
        #pragma unroll
        for (int kc = 0; kc < 4; ++kc) {
            v8h lo = *(const v8h*)&hf16[lm * 128 + kc * 32 + khA];
            v8h up = *(const v8h*)&hf16[lm * 128 + kc * 32 + 8 + khA];
            v16h a;
            #pragma unroll
            for (int e = 0; e < 8; ++e) { a[e] = lo[e]; a[8 + e] = up[e]; }
            A[kc] = a;
        }

        // ---- gh = h @ Whh^T via WMMA (Whh resident in VGPRs) -------------
        v8f z = {};
        v8f a0 = z, a1 = z, a2 = z;
        #pragma unroll
        for (int kc = 0; kc < 4; ++kc) {
            a0 = __builtin_amdgcn_wmma_f32_16x16x32_f16(false, A[kc], false, B0[kc], (short)0, a0, false, false);
            a1 = __builtin_amdgcn_wmma_f32_16x16x32_f16(false, A[kc], false, B1[kc], (short)0, a1, false, false);
        }
        if (has2) {   // wave-uniform branch: EXEC stays all-ones
            #pragma unroll
            for (int kc = 0; kc < 4; ++kc)
                a2 = __builtin_amdgcn_wmma_f32_16x16x32_f16(false, A[kc], false, B2[kc], (short)0, a2, false, false);
        }

        // ---- scatter gh to LDS -------------------------------------------
        #pragma unroll
        for (int r = 0; r < 8; ++r) {
            const int mrow = (hiL ? 8 : 0) + r;
            ghbuf[mrow * 300 + nt0 * 16 + lm] = a0[r];
            ghbuf[mrow * 300 + nt1 * 16 + lm] = a1[r];
        }
        if (has2) {
            #pragma unroll
            for (int r = 0; r < 8; ++r) {
                const int mrow = (hiL ? 8 : 0) + r;
                const int n = nt2 * 16 + lm;
                if (n < 300) ghbuf[mrow * 300 + n] = a2[r];
            }
        }
        __syncthreads();

        // ---- gate math + h update ----------------------------------------
        for (int idx = tid; idx < 16 * HIDR; idx += 256) {
            const int bl = idx / HIDR;
            const int j  = idx - bl * HIDR;
            const float* gi = gi1 + ((size_t)t * BATCH + (bglob0 + bl)) * 300;
            if (t + 1 < LLEN) __builtin_prefetch(gi + (size_t)BATCH * 300 + j, 0, 3);
            const float gr  = gi[j]       + ghbuf[bl * 300 + j]       + bhh[j];
            const float gz  = gi[100 + j] + ghbuf[bl * 300 + 100 + j] + bhh[100 + j];
            const float hn  = ghbuf[bl * 300 + 200 + j] + bhh[200 + j];
            const float r   = sigf(gr);
            const float zz  = sigf(gz);
            const float n   = tanhf(gi[200 + j] + r * hn);
            const float hp  = hbuf[bl * HIDR + j];
            const float hnew = (1.0f - zz) * n + zz * hp;
            hbuf[bl * HIDR + j] = hnew;
            hf16[bl * 128 + j]  = (_Float16)hnew;
        }
        __syncthreads();
    }

    for (int idx = tid; idx < 16 * HIDR; idx += 256) {
        const int bl = idx / HIDR;
        const int j  = idx - bl * HIDR;
        h_out[(size_t)(bglob0 + bl) * HIDR + j] = hbuf[idx];
    }
}

// ---------------------------------------------------------------------------
// Kernel 6: GRU-skip recurrence (3072 independent rows, T=6, H=5)
// ---------------------------------------------------------------------------
__global__ __launch_bounds__(256) void grus_kernel(
        const float* __restrict__ gis,   // [6][3072][15] (includes bih)
        const float* __restrict__ Whh,   // [15][5]
        const float* __restrict__ bhh,   // [15]
        float* __restrict__ hsflat)      // [128][120]
{
    __shared__ float Wsh[15 * 5];
    __shared__ float bsh[15];
    const int tid = threadIdx.x;
    if (tid < 75) Wsh[tid] = Whh[tid];
    if (tid < 15) bsh[tid] = bhh[tid];
    __syncthreads();

    const int row = blockIdx.x * 256 + tid;
    if (row >= BATCH * SKIP) return;
    const int b  = row / SKIP;
    const int sk = row % SKIP;

    float h[HIDS] = {0.f, 0.f, 0.f, 0.f, 0.f};
    for (int t = 0; t < PT; ++t) {
        const float* gi = gis + ((size_t)t * BATCH * SKIP + row) * 15;
        float hnew[HIDS];
        #pragma unroll
        for (int u = 0; u < HIDS; ++u) {
            float ghr = bsh[u], ghz = bsh[5 + u], ghn = bsh[10 + u];
            #pragma unroll
            for (int k = 0; k < HIDS; ++k) {
                ghr += h[k] * Wsh[u * HIDS + k];
                ghz += h[k] * Wsh[(5 + u) * HIDS + k];
                ghn += h[k] * Wsh[(10 + u) * HIDS + k];
            }
            const float r = sigf(gi[u] + ghr);
            const float z = sigf(gi[5 + u] + ghz);
            const float n = tanhf(gi[10 + u] + r * ghn);
            hnew[u] = (1.0f - z) * n + z * h[u];
        }
        #pragma unroll
        for (int u = 0; u < HIDS; ++u) h[u] = hnew[u];
    }
    #pragma unroll
    for (int u = 0; u < HIDS; ++u)
        hsflat[(size_t)b * (SKIP * HIDS) + sk * HIDS + u] = h[u];
}

// ---------------------------------------------------------------------------
// Kernel 7: final linear + highway + sigmoid
// ---------------------------------------------------------------------------
__global__ __launch_bounds__(256) void final_kernel(
        const float* __restrict__ h1,      // [128][100]
        const float* __restrict__ hsflat,  // [128][120]
        const float* __restrict__ lin1_w,  // [24][220]
        const float* __restrict__ lin1_b,  // [24]
        const float* __restrict__ x,       // [128][168][24]
        const float* __restrict__ hw_w,    // [24]
        const float* __restrict__ hw_b,    // [1]
        float* __restrict__ out)           // [128][24]
{
    const int id = blockIdx.x * 256 + threadIdx.x;
    if (id >= BATCH * MOUT) return;
    const int b = id / MOUT;
    const int m = id % MOUT;

    float acc = lin1_b[m];
    const float* wr = lin1_w + m * (HIDR + SKIP * HIDS);
    #pragma unroll 4
    for (int j = 0; j < HIDR; ++j)
        acc += h1[(size_t)b * HIDR + j] * wr[j];
    #pragma unroll 4
    for (int q = 0; q < SKIP * HIDS; ++q)
        acc += hsflat[(size_t)b * (SKIP * HIDS) + q] * wr[HIDR + q];

    float zacc = hw_b[0];
    #pragma unroll
    for (int wj = 0; wj < HW; ++wj)
        zacc += x[((size_t)b * PDIM + (PDIM - HW) + wj) * MOUT + m] * hw_w[wj];

    out[id] = sigf(acc + zacc);
}

// ---------------------------------------------------------------------------
// Host launcher
// ---------------------------------------------------------------------------
extern "C" void kernel_launch(void* const* d_in, const int* in_sizes, int n_in,
                              void* d_out, int out_size, void* d_ws, size_t ws_size,
                              hipStream_t stream) {
    const float* x         = (const float*)d_in[0];
    const float* conv_w    = (const float*)d_in[1];
    const float* conv_b    = (const float*)d_in[2];
    const float* gru1_Wih  = (const float*)d_in[3];
    const float* gru1_Whh  = (const float*)d_in[4];
    const float* gru1_bih  = (const float*)d_in[5];
    const float* gru1_bhh  = (const float*)d_in[6];
    const float* grus_Wih  = (const float*)d_in[7];
    const float* grus_Whh  = (const float*)d_in[8];
    const float* grus_bih  = (const float*)d_in[9];
    const float* grus_bhh  = (const float*)d_in[10];
    const float* lin1_w    = (const float*)d_in[11];
    const float* lin1_b    = (const float*)d_in[12];
    const float* hw_w      = (const float*)d_in[13];
    const float* hw_b      = (const float*)d_in[14];
    float* out = (float*)d_out;

    // workspace partition (floats); total ~45.4 MB
    float* ws  = (float*)d_ws;
    float* C1  = ws;                                   // 128*32256   = 4,128,768
    float* c2  = C1  + (size_t)BATCH * ODIM;           // 128*32*163  =   667,648
    float* gi1 = c2  + (size_t)BATCH * HIDC * LLEN;    // 163*128*300 = 6,259,200
    float* gis = gi1 + (size_t)LLEN * BATCH * 300;     // 6*3072*15   =   276,480
    float* h1  = gis + (size_t)PT * BATCH * SKIP * 15; // 128*100     =    12,800
    float* hs  = h1  + (size_t)BATCH * HIDR;           // 128*120     =    15,360

    conv_gemm_kernel<<<ODIM / 128, 256, 0, stream>>>(x, conv_w, conv_b, C1);

    window_sum_kernel<<<(BATCH * HIDC * LLEN + 255) / 256, 256, 0, stream>>>(C1, c2);

    gi1_kernel<<<(LLEN * BATCH * 300 + 255) / 256, 256, 0, stream>>>(
        c2, gru1_Wih, gru1_bih, gi1);

    gis_kernel<<<(PT * BATCH * SKIP * 15 + 255) / 256, 256, 0, stream>>>(
        c2, grus_Wih, grus_bih, gis);

    gru1_kernel<<<BATCH / 16, 256, 0, stream>>>(gi1, gru1_Whh, gru1_bhh, h1);

    grus_kernel<<<(BATCH * SKIP + 255) / 256, 256, 0, stream>>>(
        gis, grus_Whh, grus_bhh, hs);

    final_kernel<<<(BATCH * MOUT + 255) / 256, 256, 0, stream>>>(
        h1, hs, lin1_w, lin1_b, x, hw_w, hw_b, out);
}
